// LorentzLinearAttention_45157286150965
// MI455X (gfx1250) — matrix operand order repaired and checked by reading, not compiled, not run on hardware
//
#include <hip/hip_runtime.h>
#include <math.h>

// ---------------- problem constants ----------------
#define B_     8
#define N_     8192
#define DL     65          // lifted dim (x0 + 64)
#define DX     64          // spatial dim
#define M_     64          // random features
#define F_     192         // (R+1)*M combined feature dim
#define CHUNK  128
#define NCHUNK (N_ / CHUNK) // 64
#define VX     80          // V padded with ones col (65) + zeros to 80
#define SCOLS  66          // stored state cols: 65 V dims + 1 normalizer
#define PSI2   0.70710678f // sqrt(1/2!)
#define EPS_   1e-4f
#define CHUNK_BYTES (CHUNK * DL * 4) // 33280, contiguous region per chunk

typedef __attribute__((ext_vector_type(16))) __bf16 v16bf;
typedef __attribute__((ext_vector_type(8)))  float  v8f;
typedef __attribute__((ext_vector_type(4)))  int    v4i;

// ---- CDNA5 async global->LDS staging (guarded: falls back to direct loads) ----
#if defined(__has_builtin)
#if __has_builtin(__builtin_amdgcn_global_load_async_to_lds_b128) && \
    __has_builtin(__builtin_amdgcn_s_wait_asynccnt)
#define HAVE_ASYNC 1
#endif
#endif

#ifdef HAVE_ASYNC
typedef __attribute__((address_space(1))) v4i* gv4_ptr; // global 16B vector ptr
typedef __attribute__((address_space(3))) v4i* lv4_ptr; // LDS 16B vector ptr
// copy `bytes` (multiple of 16) from global to LDS via the async copy unit
__device__ __forceinline__ void async_stage(const float* g, float* lds, int bytes) {
  int n16 = bytes >> 4;
  for (int i = threadIdx.x; i < n16; i += 256) {
    __builtin_amdgcn_global_load_async_to_lds_b128(
        (gv4_ptr)((const char*)g + i * 16), (lv4_ptr)((char*)lds + i * 16), 0, 0);
  }
}
__device__ __forceinline__ void async_wait() {
  __builtin_amdgcn_s_wait_asynccnt(0);
}
#endif

// CDNA5 wave32 WMMA: D(16x16,f32) = A(16x32,bf16) x B(32x16,bf16) + C
__device__ __forceinline__ v8f wmma_bf16(v16bf a, v16bf b, v8f c) {
  return __builtin_amdgcn_wmma_f32_16x16x32_bf16(false, a, false, b, (short)0, c,
                                                 false, false);
}

// A-matrix element->K mapping (contiguous, matches B fills):
// lanes 0-15: elements hold K0..15; lanes16-31: K16..31
__device__ __forceinline__ int a_kidx(int e, int kh) {
  return 16 * kh + e;
}

// featurize a chunk: proj = xs @ omega^T via WMMA, then
// sF[t][r*64+m] = c_r(x0) * exp(proj - 0.5|xs|^2)/8, c = {1, w1sign*x0, PSI2*x0^2}
__device__ __forceinline__ void featurize_proj(const __bf16* sX,  // [CHUNK][DX]
                                               const __bf16* sOm, // [DX][M_]
                                               const float* sX0, const float* sSS,
                                               __bf16* sF,        // [CHUNK][F_]
                                               float w1sign) {
  int lane = threadIdx.x & 31;
  int wave = threadIdx.x >> 5;
  int kh = lane >> 4, ln = lane & 15;
  for (int tile = wave; tile < 32; tile += 8) { // 8 row-tiles x 4 col-tiles
    int rt = tile >> 2, mt = tile & 3;
    v8f acc = {};
#pragma unroll
    for (int kk = 0; kk < 2; ++kk) { // K = 64
      int kbase = kk * 32;
      v16bf a, bb;
      const __bf16* ap = sX + (rt * 16 + ln) * DX + kbase;
#pragma unroll
      for (int e = 0; e < 16; ++e) a[e] = ap[a_kidx(e, kh)];
#pragma unroll
      for (int e = 0; e < 16; ++e)
        bb[e] = sOm[(kbase + 16 * kh + e) * M_ + (mt * 16 + ln)];
      acc = wmma_bf16(a, bb, acc);
    }
#pragma unroll
    for (int r = 0; r < 8; ++r) {
      int row = rt * 16 + r + 8 * kh;
      int m = mt * 16 + ln;
      float x0 = sX0[row];
      float phi = __expf(acc[r] - 0.5f * sSS[row]) * 0.125f; // /sqrt(M)
      sF[row * F_ + m]       = (__bf16)phi;
      sF[row * F_ + 64 + m]  = (__bf16)(w1sign * x0 * phi);
      sF[row * F_ + 128 + m] = (__bf16)(PSI2 * x0 * x0 * phi);
    }
  }
}

__device__ __forceinline__ void load_tokens(const float* g, __bf16* sX, float* sX0,
                                            float* sSS, int t) {
  const float* p = g + t * DL;
  float x0 = p[0], ss = 0.f;
  for (int d = 0; d < DX; ++d) {
    float v = p[1 + d];
    ss += v * v;
    sX[t * DX + d] = (__bf16)v;
  }
  sX0[t] = x0;
  sSS[t] = ss;
}

__device__ __forceinline__ void load_v(const float* g, __bf16* sV, int t) {
  const float* p = g + t * DL;
  for (int d = 0; d < DL; ++d) sV[t * VX + d] = (__bf16)p[d];
  sV[t * VX + 65] = (__bf16)1.0f; // ones column -> normalizer z
  for (int d = 66; d < VX; ++d) sV[t * VX + d] = (__bf16)0.0f;
}

// ---------------- phase 1: per-chunk local sums A = Kf^T [V|1] ----------------
__global__ __launch_bounds__(256) void phase1_chunk_state(
    const float* __restrict__ K, const float* __restrict__ V,
    const float* __restrict__ omega, float* __restrict__ state) {
  __shared__ __bf16 sOm[DX * M_];
  __shared__ __bf16 sX[CHUNK * DX];
  __shared__ float sX0[CHUNK], sSS[CHUNK];
  __shared__ __bf16 sKf[CHUNK * F_];
  __shared__ __bf16 sV[CHUNK * VX];
#ifdef HAVE_ASYNC
  __shared__ float sStgK[CHUNK * DL];
  __shared__ float sStgV[CHUNK * DL];
#endif

  int b = blockIdx.x / NCHUNK, c = blockIdx.x % NCHUNK;
  int t0 = c * CHUNK, tid = threadIdx.x;
  int lane = tid & 31, wave = tid >> 5, kh = lane >> 4, ln = lane & 15;

  const float* Kb = K + ((size_t)b * N_ + t0) * DL;
  const float* Vb = V + ((size_t)b * N_ + t0) * DL;

#ifdef HAVE_ASYNC
  async_stage(Kb, sStgK, CHUNK_BYTES); // HBM -> LDS via async copy unit
  async_stage(Vb, sStgV, CHUNK_BYTES);
  const float* Ksrc = sStgK;
  const float* Vsrc = sStgV;
#else
  __builtin_prefetch(Kb, 0, 0);
  __builtin_prefetch(Vb, 0, 0);
  const float* Ksrc = Kb;
  const float* Vsrc = Vb;
#endif

  for (int i = tid; i < DX * M_; i += 256) {
    int d = i / M_, m = i % M_;
    sOm[i] = (__bf16)omega[m * DX + d]; // omega^T
  }
#ifdef HAVE_ASYNC
  async_wait();
  __syncthreads();
#endif
  if (tid < CHUNK) load_tokens(Ksrc, sX, sX0, sSS, tid);
  else load_v(Vsrc, sV, tid - CHUNK);
  __syncthreads();
  featurize_proj(sX, sOm, sX0, sSS, sKf, +1.0f);
  __syncthreads();

  float* st = state + (size_t)blockIdx.x * F_ * SCOLS;
  for (int tile = wave; tile < 12 * 5; tile += 8) { // F/16 x VX/16 tiles
    int ft = tile / 5, vt = tile % 5;
    v8f acc = {};
#pragma unroll
    for (int kk = 0; kk < 4; ++kk) { // K = CHUNK tokens
      int kbase = kk * 32;
      v16bf a, bb;
#pragma unroll
      for (int e = 0; e < 16; ++e)
        a[e] = sKf[(kbase + a_kidx(e, kh)) * F_ + ft * 16 + ln]; // Kf^T
#pragma unroll
      for (int e = 0; e < 16; ++e)
        bb[e] = sV[(kbase + 16 * kh + e) * VX + vt * 16 + ln];
      acc = wmma_bf16(a, bb, acc);
    }
#pragma unroll
    for (int r = 0; r < 8; ++r) {
      int f = ft * 16 + r + 8 * kh;
      int col = vt * 16 + ln;
      if (col < SCOLS) st[f * SCOLS + col] = acc[r];
    }
  }
}

// ---------------- phase 2: exclusive prefix over chunks (in place) ------------
__global__ __launch_bounds__(256) void phase2_prefix(float* __restrict__ state) {
  const int EPB = F_ * SCOLS;
  int idx = blockIdx.x * 256 + threadIdx.x;
  if (idx >= B_ * EPB) return;
  int b = idx / EPB, e = idx % EPB;
  float* p = state + (size_t)b * NCHUNK * EPB + e;
  float run = 0.f;
  for (int c = 0; c < NCHUNK; ++c) {
    float v = p[(size_t)c * EPB];
    p[(size_t)c * EPB] = run;
    run += v;
  }
}

// ---------------- phase 3: Y = Qf S_prefix + causal(Qf Kf^T)[V|1] -------------
__global__ __launch_bounds__(256) void phase3_output(
    const float* __restrict__ Q, const float* __restrict__ K,
    const float* __restrict__ V, const float* __restrict__ omega,
    const float* __restrict__ state, float* __restrict__ out) {
  __shared__ __bf16 sOm[DX * M_];
  __shared__ __bf16 sX[CHUNK * DX];
  __shared__ float sX0[CHUNK], sSS[CHUNK];
  __shared__ __bf16 sKf[CHUNK * F_];
  __shared__ __bf16 sQf[CHUNK * F_];
  __shared__ __bf16 sV[CHUNK * VX];
  __shared__ __bf16 sS[F_ * VX];        // prefix state, bf16
  __shared__ __bf16 sSc[CHUNK * CHUNK]; // masked scores
  __shared__ float sY[CHUNK * SCOLS];   // also doubles as Q staging buffer
#ifdef HAVE_ASYNC
  __shared__ float sStgV[CHUNK * DL];
#endif

  int b = blockIdx.x / NCHUNK, c = blockIdx.x % NCHUNK;
  int t0 = c * CHUNK, tid = threadIdx.x;
  int lane = tid & 31, wave = tid >> 5, kh = lane >> 4, ln = lane & 15;

  const float* Kb = K + ((size_t)b * N_ + t0) * DL;
  const float* Qb = Q + ((size_t)b * N_ + t0) * DL;
  const float* Vb = V + ((size_t)b * N_ + t0) * DL;
  const float* stp = state + (size_t)blockIdx.x * F_ * SCOLS;

#ifdef HAVE_ASYNC
  async_stage(Vb, sStgV, CHUNK_BYTES);
  async_stage(Qb, sY, CHUNK_BYTES); // sY region reused as Q staging
  const float* Vsrc = sStgV;
  const float* Qsrc = (const float*)sY;
#else
  __builtin_prefetch(Qb, 0, 0);
  __builtin_prefetch(stp, 0, 0);
  const float* Vsrc = Vb;
  const float* Qsrc = Qb;
#endif

  for (int i = tid; i < DX * M_; i += 256) {
    int d = i / M_, m = i % M_;
    sOm[i] = (__bf16)omega[m * DX + d];
  }
  if (tid < CHUNK) load_tokens(Kb, sX, sX0, sSS, tid); // K straight from global
#ifdef HAVE_ASYNC
  async_wait();
#endif
  __syncthreads();
  if (tid >= CHUNK) load_v(Vsrc, sV, tid - CHUNK);
  featurize_proj(sX, sOm, sX0, sSS, sKf, +1.0f);
  __syncthreads();

  if (tid < CHUNK) load_tokens(Qsrc, sX, sX0, sSS, tid);
  for (int i = tid; i < F_ * VX; i += 256) {
    int f = i / VX, col = i % VX;
    sS[i] = (__bf16)((col < SCOLS) ? stp[f * SCOLS + col] : 0.f);
  }
  __syncthreads();
  featurize_proj(sX, sOm, sX0, sSS, sQf, -1.0f); // odd block carries the minus
  __syncthreads();

  // causal-masked intra-chunk scores (K = F_)
  for (int tile = wave; tile < 64; tile += 8) {
    int qt = tile >> 3, kt = tile & 7;
    v8f acc = {};
#pragma unroll
    for (int kk = 0; kk < 6; ++kk) {
      int kbase = kk * 32;
      v16bf a, bb;
      const __bf16* ap = sQf + (qt * 16 + ln) * F_ + kbase;
#pragma unroll
      for (int e = 0; e < 16; ++e) a[e] = ap[a_kidx(e, kh)];
      const __bf16* bp = sKf + (kt * 16 + ln) * F_ + kbase + 16 * kh; // Kf^T
#pragma unroll
      for (int e = 0; e < 16; ++e) bb[e] = bp[e];
      acc = wmma_bf16(a, bb, acc);
    }
#pragma unroll
    for (int r = 0; r < 8; ++r) {
      int row = qt * 16 + r + 8 * kh;
      int col = kt * 16 + ln;
      sSc[row * CHUNK + col] = (__bf16)((col <= row) ? acc[r] : 0.f);
    }
  }
  __syncthreads();

  // Y = Qf @ sS (K=F_) + sSc @ sV (K=CHUNK)
  for (int tile = wave; tile < 40; tile += 8) {
    int rt = tile / 5, vt = tile % 5;
    v8f acc = {};
#pragma unroll
    for (int kk = 0; kk < 6; ++kk) {
      int kbase = kk * 32;
      v16bf a, bb;
      const __bf16* ap = sQf + (rt * 16 + ln) * F_ + kbase;
#pragma unroll
      for (int e = 0; e < 16; ++e) a[e] = ap[a_kidx(e, kh)];
#pragma unroll
      for (int e = 0; e < 16; ++e)
        bb[e] = sS[(kbase + 16 * kh + e) * VX + vt * 16 + ln];
      acc = wmma_bf16(a, bb, acc);
    }
#pragma unroll
    for (int kk = 0; kk < 4; ++kk) {
      int kbase = kk * 32;
      v16bf a, bb;
      const __bf16* ap = sSc + (rt * 16 + ln) * CHUNK + kbase;
#pragma unroll
      for (int e = 0; e < 16; ++e) a[e] = ap[a_kidx(e, kh)];
#pragma unroll
      for (int e = 0; e < 16; ++e)
        bb[e] = sV[(kbase + 16 * kh + e) * VX + vt * 16 + ln];
      acc = wmma_bf16(a, bb, acc);
    }
#pragma unroll
    for (int r = 0; r < 8; ++r) {
      int row = rt * 16 + r + 8 * kh;
      int col = vt * 16 + ln;
      if (col < SCOLS) sY[row * SCOLS + col] = acc[r];
    }
  }
  __syncthreads();

  // normalize with stable_inv and re-project to hyperboloid
  if (tid < CHUNK) {
    int t = tid;
    float s = sY[t * SCOLS + 65];
    float inv = (fabsf(s) > EPS_) ? (1.f / s) : (s / (s * s + EPS_ * EPS_));
    float* o = out + ((size_t)b * N_ + t0 + t) * DL;
    float ssq = 0.f;
    for (int d = 1; d < DL; ++d) {
      float y = sY[t * SCOLS + d] * inv;
      ssq += y * y;
      o[d] = y;
    }
    o[0] = sqrtf(ssq + 1.0f);
  }
}

extern "C" void kernel_launch(void* const* d_in, const int* in_sizes, int n_in,
                              void* d_out, int out_size, void* d_ws, size_t ws_size,
                              hipStream_t stream) {
  (void)in_sizes; (void)n_in; (void)out_size; (void)ws_size;
  const float* Q = (const float*)d_in[0];
  const float* K = (const float*)d_in[1];
  const float* V = (const float*)d_in[2];
  const float* omega = (const float*)d_in[3];
  float* out = (float*)d_out;
  float* state = (float*)d_ws; // B*NCHUNK*F_*SCOLS f32 ~= 26 MB, fully rewritten

  dim3 blk(256);
  phase1_chunk_state<<<dim3(B_ * NCHUNK), blk, 0, stream>>>(K, V, omega, state);
  int total = B_ * F_ * SCOLS;
  phase2_prefix<<<dim3((total + 255) / 256), blk, 0, stream>>>(state);
  phase3_output<<<dim3(B_ * NCHUNK), blk, 0, stream>>>(Q, K, V, omega, state, out);
}
